// GInRecConv_62637803045254
// MI455X (gfx1250) — compile-verified, matrix operand order checked by reading, not run on hardware
//
#include <hip/hip_runtime.h>
#include <hip/hip_bf16.h>

#define NN 50000
#define NE 500000
#define NR 8
#define DD 128      // input/output dim
#define AD 64       // attention dim

typedef _Float16 v16h __attribute__((ext_vector_type(16)));
typedef _Float16 v8h  __attribute__((ext_vector_type(8)));
typedef _Float16 v4h  __attribute__((ext_vector_type(4)));
typedef float    v8f  __attribute__((ext_vector_type(8)));
typedef float    v4f  __attribute__((ext_vector_type(4)));

__device__ __forceinline__ v8f v8f_zero() {
  v8f z;
  #pragma unroll
  for (int i = 0; i < 8; ++i) z[i] = 0.f;
  return z;
}

// Load one 16x32 f16 A/B fragment slice for this lane.
// p points at element [row, k0]; layout: elems k0..k0+7 then k0+16..k0+23.
__device__ __forceinline__ v16h load_frag16(const _Float16* p) {
  union { v16h v; v8h h[2]; } u;
  u.h[0] = *(const v8h*)(p);
  u.h[1] = *(const v8h*)(p + 16);
  return u.v;
}

__device__ __forceinline__ v8f wmma_f16(v16h a, v16h b, v8f c) {
  return __builtin_amdgcn_wmma_f32_16x16x32_f16(false, a, false, b,
                                                (short)0, c, false, false);
}

__device__ __forceinline__ void atomicMaxFloat(float* addr, float val) {
  // IEEE trick: non-negative floats order as ints, negative as reversed uints.
  if (!(val < 0.f)) atomicMax((int*)addr, __float_as_int(val));
  else              atomicMin((unsigned int*)addr, __float_as_uint(val));
}

// ---------------- conversion / init kernels ----------------

__global__ void k_cvt(const float* __restrict__ in, _Float16* __restrict__ out, int n) {
  int i = blockIdx.x * blockDim.x + threadIdx.x;
  if (i < n) out[i] = (_Float16)in[i];
}

// relations [R, 2*D, D] (e contiguous) -> relT16[(half*8+r)][e][d] (d contiguous)
__global__ void k_cvt_rel(const float* __restrict__ rel, _Float16* __restrict__ relT) {
  int i = blockIdx.x * blockDim.x + threadIdx.x;   // over 16*128*128
  if (i >= 16 * DD * DD) return;
  int d = i & (DD - 1);
  int e = (i >> 7) & (DD - 1);
  int slot = i >> 14;                // half*8 + r
  int half = slot >> 3, r = slot & 7;
  relT[i] = (_Float16)rel[((size_t)r * 2 * DD + half * DD + d) * DD + e];
}

__global__ void k_init(float* __restrict__ hN, float* __restrict__ smax,
                       float* __restrict__ denom, int nD, int n) {
  int i = blockIdx.x * blockDim.x + threadIdx.x;
  if (i < nD) hN[i] = 0.f;
  if (i < n) { smax[i] = __int_as_float(0xFF800000); denom[i] = 0.f; }
}

// ---------------- relation projections (WMMA) ----------------
// For slot = half*8+r: P[slot][n][e] = sum_d h[n,d] * relations[r, half*128+d, e]
// One wave computes a 16(node) x 128(e) tile. Stored f16.
__global__ void k_proj(const _Float16* __restrict__ h16,
                       const _Float16* __restrict__ relT16,
                       _Float16* __restrict__ proj16) {
  const int mtiles = NN / 16;
  int bid = blockIdx.x;
  int mtile = bid % mtiles;
  int slot = bid / mtiles;           // 0..15
  int lane = threadIdx.x;
  int lo = lane & 15, hi = lane >> 4;

  const _Float16* A = h16 + (size_t)(mtile * 16 + lo) * DD + hi * 8;
  const _Float16* B = relT16 + ((size_t)slot * DD + lo) * DD + hi * 8;

  v8f acc[8];
  #pragma unroll
  for (int j = 0; j < 8; ++j) acc[j] = v8f_zero();

  #pragma unroll
  for (int kk = 0; kk < 4; ++kk) {
    v16h a = load_frag16(A + kk * 32);
    #pragma unroll
    for (int jt = 0; jt < 8; ++jt) {
      v16h b = load_frag16(B + (size_t)jt * 16 * DD + kk * 32);
      acc[jt] = wmma_f16(a, b, acc[jt]);
    }
  }

  _Float16* out = proj16 + (size_t)slot * NN * DD;
  #pragma unroll
  for (int jt = 0; jt < 8; ++jt) {
    int col = jt * 16 + lo;
    #pragma unroll
    for (int v = 0; v < 8; ++v) {
      int row = mtile * 16 + v + 8 * hi;
      out[(size_t)row * DD + col] = (_Float16)acc[jt][v];
    }
  }
}

// ---------------- GATv2 attention scores (WMMA, gathered A) ----------------
// a = leaky([dst_h ; src_h] @ Wa^T + b_a); score = sum_j att0[j]*a[j]
__global__ void k_attn(const _Float16* __restrict__ h16,
                       const _Float16* __restrict__ Wa16,
                       const float* __restrict__ rel_att,   // row 0 of rel_attention
                       const float* __restrict__ b_a,
                       const int* __restrict__ src, const int* __restrict__ dst,
                       float* __restrict__ score, float* __restrict__ smax) {
  int e0 = blockIdx.x * 16;
  int lane = threadIdx.x;
  int lo = lane & 15, hi = lane >> 4;
  int e = e0 + lo;
  int dn = dst[e], sn = src[e];

  v8f acc[4];
  #pragma unroll
  for (int j = 0; j < 4; ++j) acc[j] = v8f_zero();

  #pragma unroll
  for (int kk = 0; kk < 8; ++kk) {
    int k0 = kk * 32 + hi * 8;       // both 8-runs stay within one half of concat
    const _Float16* ap = (k0 < DD) ? (h16 + (size_t)dn * DD + k0)
                                   : (h16 + (size_t)sn * DD + (k0 - DD));
    v16h a = load_frag16(ap);
    #pragma unroll
    for (int jt = 0; jt < 4; ++jt) {
      v16h b = load_frag16(Wa16 + (size_t)(jt * 16 + lo) * (2 * DD) + k0);
      acc[jt] = wmma_f16(a, b, acc[jt]);
    }
  }

  float s[8];
  #pragma unroll
  for (int v = 0; v < 8; ++v) s[v] = 0.f;
  #pragma unroll
  for (int jt = 0; jt < 4; ++jt) {
    int j = jt * 16 + lo;
    float w = rel_att[j], bb = b_a[j];
    #pragma unroll
    for (int v = 0; v < 8; ++v) {
      float x = acc[jt][v] + bb;
      x = (x > 0.f) ? x : 0.01f * x;
      s[v] += x * w;
    }
  }
  // reduce across the 16 columns held by each half-wave
  #pragma unroll
  for (int v = 0; v < 8; ++v) {
    float t = s[v];
    t += __shfl_xor(t, 1, 32);
    t += __shfl_xor(t, 2, 32);
    t += __shfl_xor(t, 4, 32);
    t += __shfl_xor(t, 8, 32);
    s[v] = t;
  }
  if (lo == 0) {
    #pragma unroll
    for (int v = 0; v < 8; ++v) {
      int ee = e0 + v + 8 * hi;
      score[ee] = s[v];
      atomicMaxFloat(&smax[dst[ee]], s[v]);
    }
  }
}

// ---------------- softmax numerator + denominator ----------------
__global__ void k_exp(const float* __restrict__ score, const int* __restrict__ dst,
                      const float* __restrict__ smax, float* __restrict__ ex,
                      float* __restrict__ denom) {
  int e = blockIdx.x * blockDim.x + threadIdx.x;
  if (e >= NE) return;
  int dn = dst[e];
  float v = __expf(score[e] - smax[dn]);
  ex[e] = v;
  atomicAdd(&denom[dn], v);
}

// ---------------- gated message scatter (one wave per edge) ----------------
__global__ void k_msg(const float* __restrict__ h,
                      const _Float16* __restrict__ proj16,
                      const int* __restrict__ src, const int* __restrict__ dst,
                      const int* __restrict__ etype,
                      const float* __restrict__ ex, const float* __restrict__ denom,
                      float* __restrict__ hN) {
  int e = blockIdx.x * (blockDim.x >> 5) + (threadIdx.x >> 5);
  int lane = threadIdx.x & 31;
  if (e >= NE) return;
  int dn = dst[e], sn = src[e], et = etype[e];
  float attn = ex[e] / denom[dn];
  const _Float16* pd = proj16 + ((size_t)et * NN + dn) * DD + lane * 4;       // dst half
  const _Float16* ps = proj16 + ((size_t)(8 + et) * NN + sn) * DD + lane * 4; // src half
  v4h pdv = *(const v4h*)pd;
  v4h psv = *(const v4h*)ps;
  v4f hs = *(const v4f*)(h + (size_t)sn * DD + lane * 4);
  float* o = hN + (size_t)dn * DD + lane * 4;
  #pragma unroll
  for (int i = 0; i < 4; ++i) {
    float g = (float)pdv[i] + (float)psv[i];
    g = 1.f / (1.f + __expf(-g));
    atomicAdd(o + i, hs[i] * attn * g);
  }
}

// ---------------- final linear: leaky(W_lin [h ; h_N] + b_lin) (WMMA) ----------------
__global__ void k_final(const _Float16* __restrict__ h16,
                        const float* __restrict__ hN,
                        const _Float16* __restrict__ Wl16,
                        const float* __restrict__ b_lin,
                        float* __restrict__ out) {
  int mtile = blockIdx.x;
  int lane = threadIdx.x;
  int lo = lane & 15, hi = lane >> 4;
  int row = mtile * 16 + lo;

  v8f acc[8];
  #pragma unroll
  for (int j = 0; j < 8; ++j) acc[j] = v8f_zero();

  #pragma unroll
  for (int kk = 0; kk < 8; ++kk) {
    int k0 = kk * 32 + hi * 8;
    v16h a;
    if (k0 < DD) {
      a = load_frag16(h16 + (size_t)row * DD + k0);
    } else {
      const float* p = hN + (size_t)row * DD + (k0 - DD);
      union { v16h v; v4h h4[4]; } u;
      v4f f0 = *(const v4f*)(p);
      v4f f1 = *(const v4f*)(p + 4);
      v4f f2 = *(const v4f*)(p + 16);
      v4f f3 = *(const v4f*)(p + 20);
      #pragma unroll
      for (int i = 0; i < 4; ++i) {
        u.h4[0][i] = (_Float16)f0[i];
        u.h4[1][i] = (_Float16)f1[i];
        u.h4[2][i] = (_Float16)f2[i];
        u.h4[3][i] = (_Float16)f3[i];
      }
      a = u.v;
    }
    #pragma unroll
    for (int jt = 0; jt < 8; ++jt) {
      v16h b = load_frag16(Wl16 + (size_t)(jt * 16 + lo) * (2 * DD) + k0);
      acc[jt] = wmma_f16(a, b, acc[jt]);
    }
  }

  #pragma unroll
  for (int jt = 0; jt < 8; ++jt) {
    int col = jt * 16 + lo;
    float bb = b_lin[col];
    #pragma unroll
    for (int v = 0; v < 8; ++v) {
      float x = acc[jt][v] + bb;
      x = (x > 0.f) ? x : 0.01f * x;
      out[(size_t)(mtile * 16 + v + 8 * hi) * DD + col] = x;
    }
  }
}

extern "C" void kernel_launch(void* const* d_in, const int* in_sizes, int n_in,
                              void* d_out, int out_size, void* d_ws, size_t ws_size,
                              hipStream_t stream) {
  const float* h         = (const float*)d_in[0];
  const float* relations = (const float*)d_in[1];
  const float* rel_att   = (const float*)d_in[2];   // [8,64]; reference uses row 0
  const float* W_a       = (const float*)d_in[3];
  const float* b_a       = (const float*)d_in[4];
  const float* W_lin     = (const float*)d_in[5];
  const float* b_lin     = (const float*)d_in[6];
  const int*   src       = (const int*)d_in[7];
  const int*   dst       = (const int*)d_in[8];
  const int*   etype     = (const int*)d_in[9];
  float* out = (float*)d_out;
  (void)in_sizes; (void)n_in; (void)out_size; (void)ws_size;

  char* ws = (char*)d_ws;
  size_t off = 0;
  auto take = [&](size_t bytes) -> char* {
    char* p = ws + off;
    off = (off + bytes + 255) & ~(size_t)255;
    return p;
  };
  _Float16* h16    = (_Float16*)take((size_t)NN * DD * 2);
  _Float16* relT16 = (_Float16*)take((size_t)16 * DD * DD * 2);
  _Float16* Wa16   = (_Float16*)take((size_t)AD * 2 * DD * 2);
  _Float16* Wl16   = (_Float16*)take((size_t)DD * 2 * DD * 2);
  _Float16* proj16 = (_Float16*)take((size_t)16 * NN * DD * 2);   // [2*R][N][D] f16
  float* score = (float*)take((size_t)NE * 4);
  float* ex    = (float*)take((size_t)NE * 4);
  float* smax  = (float*)take((size_t)NN * 4);
  float* denom = (float*)take((size_t)NN * 4);
  float* hN    = (float*)take((size_t)NN * DD * 4);

  const int nh = NN * DD;
  k_cvt<<<(nh + 255) / 256, 256, 0, stream>>>(h, h16, nh);
  k_cvt<<<(AD * 2 * DD + 255) / 256, 256, 0, stream>>>(W_a, Wa16, AD * 2 * DD);
  k_cvt<<<(DD * 2 * DD + 255) / 256, 256, 0, stream>>>(W_lin, Wl16, DD * 2 * DD);
  k_cvt_rel<<<(16 * DD * DD) / 256, 256, 0, stream>>>(relations, relT16);
  k_init<<<(nh + 255) / 256, 256, 0, stream>>>(hN, smax, denom, nh, NN);

  k_proj<<<(NN / 16) * 16, 32, 0, stream>>>(h16, relT16, proj16);
  k_attn<<<NE / 16, 32, 0, stream>>>(h16, Wa16, rel_att, b_a, src, dst, score, smax);
  k_exp<<<(NE + 255) / 256, 256, 0, stream>>>(score, dst, smax, ex, denom);
  k_msg<<<(NE + 7) / 8, 256, 0, stream>>>(h, proj16, src, dst, etype, ex, denom, hN);
  k_final<<<NN / 16, 32, 0, stream>>>(h16, hN, Wl16, b_lin, out);
}
